// DynaResidualBlock_36953898615316
// MI455X (gfx1250) — matrix operand order repaired and checked by reading, not compiled, not run on hardware
//
#include <hip/hip_runtime.h>
#include <hip/hip_bf16.h>

// ---- types for WMMA / TDM operands ----
typedef __attribute__((ext_vector_type(2)))  float        v2f;
typedef __attribute__((ext_vector_type(8)))  float        v8f;
typedef __attribute__((ext_vector_type(16))) _Float16     v16h;
typedef __attribute__((ext_vector_type(4)))  unsigned int u32x4;
typedef __attribute__((ext_vector_type(8)))  int          i32x8;
typedef __attribute__((ext_vector_type(4)))  int          i32x4;

#define HW   16384   // 128*128
#define B_   32
#define FIN  32
#define FOUT 32
#define FH   64
#define KTOT 9408

// ============================================================================
// Hypernetwork GEMM:  Y[32,N] = act( A[32,K] @ W[N,K]^T + bias )  (opt +=)
// f32 WMMA 16x16x4. Block = 4 waves; each wave owns one 16-wide N tile and
// both 16-row M tiles. W/A staged through LDS with coalesced b128 loads
// (row stride padded to 36 floats -> conflict-free column reads).
// Requires N % 64 == 0 and K % 32 == 0 (true for all layers here).
// ============================================================================
#define LDP 36   // padded row stride (floats): 16B-aligned rows, no bank conflicts

__global__ __launch_bounds__(128)
void hyper_gemm(const float* __restrict__ A,   // [32,K]
                const float* __restrict__ W,   // [N,K]
                const float* __restrict__ bias,// [N]
                float* __restrict__ Y,         // [32,N]
                int K, int N, int do_relu, int accumulate)
{
    __shared__ float sA[32 * LDP];        // A chunk: 32 rows x 32 k (padded)
    __shared__ float sW[4][16 * LDP];     // per-wave W chunk: 16 rows x 32 k

    const int tid   = threadIdx.x;
    const int lane  = tid & 31;
    const int wave  = tid >> 5;
    const int ntile = blockIdx.x * 4 + wave;
    const int nrow  = ntile * 16;
    const int n     = nrow + (lane & 15);
    const int khalf = lane >> 4;          // half-wave: 0 or 1
    const int m0    = lane & 15;          // row within a 16-row M tile

    v8f acc0 = {}; v8f acc1 = {};
    if (accumulate) {
#pragma unroll
        for (int r = 0; r < 8; ++r) {
            const int m = r + khalf * 8;               // D-layout row mapping
            acc0[r] = Y[(size_t)m * N + n];
            acc1[r] = Y[(size_t)(m + 16) * N + n];
        }
    }

    for (int k0 = 0; k0 < K; k0 += 32) {
        // ---- cooperative A chunk load: 32x32 floats, 128 thr x 8 (2x b128) ----
        {
            const int m  = tid >> 2;                   // 4 threads per row
            const int kc = (tid & 3) * 8;              // 8 floats each
            const float4* src = (const float4*)(A + (size_t)m * K + k0 + kc);
            float4 u0 = src[0], u1 = src[1];
            *(float4*)&sA[m * LDP + kc]     = u0;
            *(float4*)&sA[m * LDP + kc + 4] = u1;
        }
        // ---- per-wave W chunk load: 16x32 floats, 32 lanes x 16 (4x b128) ----
        {
            const int r  = lane >> 1;                  // 2 lanes per row
            const int kc = (lane & 1) * 16;            // 16 floats each
            const float4* src = (const float4*)(W + (size_t)(nrow + r) * K + k0 + kc);
            float4 u0 = src[0], u1 = src[1], u2 = src[2], u3 = src[3];
            float* dst = &sW[wave][r * LDP + kc];
            *(float4*)(dst + 0)  = u0;
            *(float4*)(dst + 4)  = u1;
            *(float4*)(dst + 8)  = u2;
            *(float4*)(dst + 12) = u3;
        }
        __syncthreads();

#pragma unroll
        for (int ks4 = 0; ks4 < 32; ks4 += 4) {
            const int kk = ks4 + khalf * 2;  // A 16x4 f32: lanes0-15 K0,K1 ; lanes16-31 K2,K3
            v2f a0, a1, bb;
            a0.x = sA[m0 * LDP + kk];        a0.y = sA[m0 * LDP + kk + 1];
            a1.x = sA[(m0 + 16) * LDP + kk]; a1.y = sA[(m0 + 16) * LDP + kk + 1];
            bb.x = sW[wave][(lane & 15) * LDP + kk];
            bb.y = sW[wave][(lane & 15) * LDP + kk + 1];
            acc0 = __builtin_amdgcn_wmma_f32_16x16x4_f32(false, a0, false, bb, (short)0, acc0, false, false);
            acc1 = __builtin_amdgcn_wmma_f32_16x16x4_f32(false, a1, false, bb, (short)0, acc1, false, false);
        }
        __syncthreads();
    }

    const float bv = bias ? bias[n] : 0.0f;
#pragma unroll
    for (int r = 0; r < 8; ++r) {
        const int m = r + khalf * 8;
        float v0 = acc0[r] + bv;
        float v1 = acc1[r] + bv;
        if (do_relu) { v0 = fmaxf(v0, 0.0f); v1 = fmaxf(v1, 0.0f); }
        Y[(size_t)m * N + n]        = v0;
        Y[(size_t)(m + 16) * N + n] = v1;
    }
}

// ============================================================================
// Fused per-sample dynamic 1x1-conv chain, f16 WMMA (f32 accum):
//   out = k_out @ relu(k_mid @ relu(k_in @ x + b_in) + b_mid) + b_out
//         + k_short @ x + b_short
// Generated params DMA'd to LDS via the Tensor Data Mover (TENSOR_LOAD_TO_LDS,
// completion via s_wait_tensorcnt), converted once to f16, then each wave does
// 18 f16 WMMAs per 16-pixel tile.
// ============================================================================

// f16 A-operand (16x32) from row-major LDS matrix, ld = K width.
// Layout: lanes0-15 hold K 0-15, lanes16-31 hold K 16-31 (matches B operands).
__device__ inline v16h lds_a(const _Float16* m, int ld, int row, int kb, int hi) {
    v16h a;
#pragma unroll
    for (int j = 0; j < 16; ++j) {
        const int k = kb + hi * 16 + j;
        a[j] = m[row * ld + k];
    }
    return a;
}

// f16 B-operand (32x16) from per-wave stage buffer [ch][pix].
// Lanes0-15: K 0-15 (hi=0 base), lanes16-31: K 16-31.
__device__ inline v16h lds_b(const _Float16* st, int chbase, int pix) {
    v16h b;
#pragma unroll
    for (int j = 0; j < 16; ++j) b[j] = st[(chbase + j) * 16 + pix];
    return b;
}

__global__ __launch_bounds__(256)
void dyna_conv(const float* __restrict__ x,    // [32,32,128,128]
               const float* __restrict__ ks,   // [32,9408]
               float* __restrict__ out)        // [32,32,128,128]
{
    // Raw f32 staging for the TDM copy; reused as per-wave D->B restage buffers
    // (8 waves x 64ch x 16pix f16 = 16 KB < 37632 B) after conversion.
    __shared__ float    s_raw[KTOT];        // 37632 B
    __shared__ _Float16 s_k[9216];          // k_in 2048 | k_mid 4096 | k_out 2048 | k_short 1024
    __shared__ float    s_b[192];           // b_in 64 | b_mid 64 | b_out 32 | b_short 32

    const int b   = blockIdx.x;
    const int tid = threadIdx.x;
    const int lane = tid & 31;
    const int wave = tid >> 5;
    const float* kb = ks + (size_t)b * KTOT;

#if __has_builtin(__builtin_amdgcn_tensor_load_to_lds)
    // ---- TDM: 1-D tensor copy ks[b, 0:9408] (f32) -> LDS s_raw ----
    if (wave == 0) {
        const unsigned lds_off = (unsigned)(unsigned long long)(void*)s_raw;
        const unsigned long long ga = (unsigned long long)kb;
        u32x4 g0;
        g0[0] = 1u;                                              // count=1, user D#
        g0[1] = lds_off;                                         // lds_addr
        g0[2] = (unsigned)(ga & 0xFFFFFFFFu);                    // global_addr[31:0]
        g0[3] = (unsigned)((ga >> 32) & 0x1FFFFFFu) | (2u << 30);// global_addr[56:32] | type=2
        i32x8 g1;
        g1[0] = 0x00020000;                  // workgroup_mask=0, data_size=2 (4B)
        g1[1] = (int)((KTOT & 0xFFFF) << 16);//   tensor_dim0[15:0] in [31:16]
        g1[2] = (int)((KTOT >> 16) | (1 << 16)); // tensor_dim0[31:16]; tensor_dim1=1
        g1[3] = (int)(KTOT << 16);           // tensor_dim1 hi=0 | tile_dim0=9408
        g1[4] = 0;                           // tile_dim1=0, tile_dim2=0 (unused)
        g1[5] = KTOT;                        // tensor_dim0_stride low32
        g1[6] = 0;
        g1[7] = 0;
        i32x4 z4; z4[0] = 0; z4[1] = 0; z4[2] = 0; z4[3] = 0;    // groups 2/3 unused (<=2D)
        i32x8 z8;
        z8[0] = 0; z8[1] = 0; z8[2] = 0; z8[3] = 0;
        z8[4] = 0; z8[5] = 0; z8[6] = 0; z8[7] = 0;
        __builtin_amdgcn_tensor_load_to_lds(g0, g1, z4, z4, z8, 0);
        __builtin_amdgcn_s_wait_tensorcnt(0);
    }
    __syncthreads();
    for (int i = tid; i < 9216; i += 256) s_k[i] = (_Float16)s_raw[i];
    for (int i = tid; i < 192;  i += 256) s_b[i] = s_raw[9216 + i];
#else
    for (int i = tid; i < 9216; i += 256) s_k[i] = (_Float16)kb[i];
    for (int i = tid; i < 192;  i += 256) s_b[i] = kb[9216 + i];
#endif
    __syncthreads();   // conversion done; s_raw now reusable as stage buffers

    const int pix  = lane & 15;
    const int hi   = lane >> 4;
    const int mrow = lane & 15;
    _Float16* st = ((_Float16*)s_raw) + wave * (64 * 16);

    const float* xb = x + (size_t)b * FIN * HW;

#pragma unroll 1
    for (int i = 0; i < 8; ++i) {
        const int tile = blockIdx.y * 64 + i * 8 + wave;   // 1024 tiles / sample
        const int p    = tile * 16 + pix;

        // prefetch next tile's pixels for this lane's channel slice
        __builtin_prefetch(xb + (size_t)(hi * 16) * HW + p + 16, 0, 0);

        // ---- x B-operand: 32 channels x 16 pixels, f32 -> f16 ----
        v16h bx;
#pragma unroll
        for (int j = 0; j < 16; ++j)
            bx[j] = (_Float16)xb[(size_t)(hi * 16 + j) * HW + p];

        // ---- layer 1: h1 = relu(k_in[64x32] @ x + b_in), stage to LDS ----
#pragma unroll
        for (int mt = 0; mt < 4; ++mt) {
            v16h a = lds_a(s_k + 0, 32, mt * 16 + mrow, 0, hi);
            v8f c = {};
            c = __builtin_amdgcn_wmma_f32_16x16x32_f16(false, a, false, bx, (short)0, c, false, false);
#pragma unroll
            for (int r = 0; r < 8; ++r) {
                const int ch = mt * 16 + r + hi * 8;
                st[ch * 16 + pix] = (_Float16)fmaxf(c[r] + s_b[ch], 0.0f);
            }
        }

        // ---- layer 2: h2 = relu(k_mid[64x64] @ h1 + b_mid) ----
        v16h bm0 = lds_b(st, hi * 16, pix);        // K 0-31
        v16h bm1 = lds_b(st, 32 + hi * 16, pix);   // K 32-63
#pragma unroll
        for (int mt = 0; mt < 4; ++mt) {
            v16h a0 = lds_a(s_k + 2048, 64, mt * 16 + mrow, 0,  hi);
            v16h a1 = lds_a(s_k + 2048, 64, mt * 16 + mrow, 32, hi);
            v8f c = {};
            c = __builtin_amdgcn_wmma_f32_16x16x32_f16(false, a0, false, bm0, (short)0, c, false, false);
            c = __builtin_amdgcn_wmma_f32_16x16x32_f16(false, a1, false, bm1, (short)0, c, false, false);
#pragma unroll
            for (int r = 0; r < 8; ++r) {
                const int ch = mt * 16 + r + hi * 8;
                st[ch * 16 + pix] = (_Float16)fmaxf(c[r] + s_b[64 + ch], 0.0f);
            }
        }

        // ---- layer 3: out = k_out[32x64] @ h2 + k_short[32x32] @ x + biases ----
        v16h bo0 = lds_b(st, hi * 16, pix);
        v16h bo1 = lds_b(st, 32 + hi * 16, pix);
#pragma unroll
        for (int mt = 0; mt < 2; ++mt) {
            v16h a0 = lds_a(s_k + 6144, 64, mt * 16 + mrow, 0,  hi);
            v16h a1 = lds_a(s_k + 6144, 64, mt * 16 + mrow, 32, hi);
            v16h as = lds_a(s_k + 8192, 32, mt * 16 + mrow, 0,  hi);
            v8f c = {};
            c = __builtin_amdgcn_wmma_f32_16x16x32_f16(false, a0, false, bo0, (short)0, c, false, false);
            c = __builtin_amdgcn_wmma_f32_16x16x32_f16(false, a1, false, bo1, (short)0, c, false, false);
            c = __builtin_amdgcn_wmma_f32_16x16x32_f16(false, as, false, bx,  (short)0, c, false, false);
#pragma unroll
            for (int r = 0; r < 8; ++r) {
                const int ch = mt * 16 + r + hi * 8;
                out[((size_t)b * FOUT + ch) * HW + p] = c[r] + s_b[128 + ch] + s_b[160 + ch];
            }
        }
    }
}

// ============================================================================
extern "C" void kernel_launch(void* const* d_in, const int* in_sizes, int n_in,
                              void* d_out, int out_size, void* d_ws, size_t ws_size,
                              hipStream_t stream) {
    const float* x   = (const float*)d_in[0];
    const float* lat = (const float*)d_in[1];
    const float* w0  = (const float*)d_in[2];
    const float* b0  = (const float*)d_in[3];
    const float* w1a = (const float*)d_in[4];
    const float* b1a = (const float*)d_in[5];
    const float* w1b = (const float*)d_in[6];
    const float* b1b = (const float*)d_in[7];
    const float* w1s = (const float*)d_in[8];
    const float* b1s = (const float*)d_in[9];
    const float* w2a = (const float*)d_in[10];
    const float* b2a = (const float*)d_in[11];
    const float* w2b = (const float*)d_in[12];
    const float* b2b = (const float*)d_in[13];
    const float* w2s = (const float*)d_in[14];
    const float* b2s = (const float*)d_in[15];
    const float* wfa = (const float*)d_in[16];
    const float* bfa = (const float*)d_in[17];

    float* ws  = (float*)d_ws;
    float* h0  = ws;                        // 32*512
    float* t1  = ws + 16384;                // 32*1024
    float* h1  = ws + 49152;                // 32*512
    float* t2  = ws + 65536;                // 32*1024
    float* h2  = ws + 98304;                // 32*9408
    float* ksb = ws + 98304 + 301056;       // 32*9408

    auto gemm = [&](const float* A, const float* W, const float* Bv, float* Y,
                    int K, int N, int relu, int acc) {
        const int blocks = N / 64;          // 4 N-tiles (waves) per block
        hyper_gemm<<<dim3(blocks), dim3(128), 0, stream>>>(A, W, Bv, Y, K, N, relu, acc);
    };

    // h0 = lat @ w0^T + b0
    gemm(lat, w0, b0, h0, 512, 512, 0, 0);
    // LRB1: h1 = h0@w1s^T + b1s + relu(h0@w1a^T + b1a)@w1b^T + b1b
    gemm(h0, w1a, b1a, t1, 512, 1024, 1, 0);
    gemm(h0, w1s, b1s, h1, 512, 512, 0, 0);
    gemm(t1, w1b, b1b, h1, 1024, 512, 0, 1);
    // LRB2 -> [32, 9408]
    gemm(h1, w2a, b2a, t2, 512, 1024, 1, 0);
    gemm(h1, w2s, b2s, h2, 512, KTOT, 0, 0);
    gemm(t2, w2b, b2b, h2, 1024, KTOT, 0, 1);
    // ks = h2 @ wfa^T + bfa   (354 MB weight read -> bandwidth bound)
    gemm(h2, wfa, bfa, ksb, KTOT, KTOT, 0, 0);

    // fused per-sample dynamic conv chain
    dyna_conv<<<dim3(B_, 16), dim3(256), 0, stream>>>(x, ksb, (float*)d_out);
}
